// Block_72138270704025
// MI455X (gfx1250) — matrix operand order repaired and checked by reading, not compiled, not run on hardware
//
#include <hip/hip_runtime.h>
#include <math.h>
#include <stdint.h>

// ---------------- types ----------------
typedef __bf16 v8bf  __attribute__((ext_vector_type(8)));
typedef __bf16 v16bf __attribute__((ext_vector_type(16)));
typedef float  v8f   __attribute__((ext_vector_type(8)));
typedef unsigned int u32x4 __attribute__((ext_vector_type(4)));
typedef unsigned int u32x8 __attribute__((ext_vector_type(8)));

// Problem constants (from reference)
#define DIMC   768
#define NHEAD  12
#define HD     64
#define WS     14
#define NTOK   196          // WS*WS
#define TOKP   224          // padded to 14*16 (and 7*32 k-chunks)
#define NWIN   200          // B(8) * 25 windows
#define MWIN   (NWIN*TOKP)  // 44800 rows, = 350*128
#define MTOK   32768        // 8*64*64, = 256*128

__device__ inline v16bf join8(v8bf lo, v8bf hi) {
    return __builtin_shufflevector(lo, hi, 0,1,2,3,4,5,6,7,8,9,10,11,12,13,14,15);
}
__device__ inline v8f wmma_bf16(v16bf a, v16bf b, v8f c) {
    // D = A(16x32) * B(32x16) + C, fp32 accum
    return __builtin_amdgcn_wmma_f32_16x16x32_bf16(false, a, false, b, (short)0, c, false, false);
}
__device__ inline void wait_tensorcnt0() {
#if defined(__has_builtin)
#if __has_builtin(__builtin_amdgcn_s_wait_tensorcnt)
    __builtin_amdgcn_s_wait_tensorcnt(0);
#else
    asm volatile("s_wait_tensorcnt 0" ::: "memory");
#endif
#else
    asm volatile("s_wait_tensorcnt 0" ::: "memory");
#endif
}

// ---------------- small prep kernels ----------------
__global__ void cvt_bf16_kernel(const float* __restrict__ src, __bf16* __restrict__ dst, int n) {
    int i = blockIdx.x * 256 + threadIdx.x;
    if (i < n) dst[i] = (__bf16)src[i];
}

// Sh[i,j] = sum_c rel_h[(i-j+13)*64 + c]; same for Sw.
__global__ void rel_table_kernel(const float* __restrict__ rel_h, const float* __restrict__ rel_w,
                                 float* __restrict__ Sh, float* __restrict__ Sw) {
    int idx = blockIdx.x * 256 + threadIdx.x;
    if (idx >= 2 * NTOK) return;
    const float* rp = (idx < NTOK) ? rel_h : rel_w;
    float*       op = (idx < NTOK) ? Sh : Sw;
    int t = idx % NTOK, i = t / WS, j = t % WS;
    const float* row = rp + (size_t)(i - j + WS - 1) * HD;
    float s = 0.f;
    #pragma unroll 8
    for (int c = 0; c < HD; c++) s += row[c];
    op[t] = s;
}

// ---------------- layernorm helpers/kernels ----------------
__device__ inline float block_sum256(float v, float* red) {
    int tid = threadIdx.x;
    red[tid] = v; __syncthreads();
    #pragma unroll
    for (int s = 128; s > 0; s >>= 1) {
        if (tid < s) red[tid] += red[tid + s];
        __syncthreads();
    }
    float r = red[0]; __syncthreads();
    return r;
}

// LN1 fused with window partition: hwin[win][tok(224)][768] bf16; pad rows -> 0
__global__ __launch_bounds__(256) void ln1_win_kernel(const float* __restrict__ x,
                                                      const float* __restrict__ w,
                                                      const float* __restrict__ b,
                                                      __bf16* __restrict__ hwin) {
    __shared__ float red[256];
    int wt  = blockIdx.x;            // 0..44799
    int win = wt / TOKP, tok = wt - win * TOKP;
    int bb  = win / 25, wr = win - bb * 25;
    int wy  = wr / 5,   wx = wr - wy * 5;
    int ty  = tok / WS, tx = tok - ty * WS;
    int y   = wy * WS + ty, xx = wx * WS + tx;
    __bf16* dst = hwin + (size_t)wt * DIMC;
    if (tok >= NTOK || y >= 64 || xx >= 64) {
        for (int c = threadIdx.x; c < DIMC; c += 256) dst[c] = (__bf16)0.0f;
        return;
    }
    const float* src = x + (((size_t)bb * 64 + y) * 64 + xx) * DIMC;
    int c = threadIdx.x;
    float v0 = src[c], v1 = src[c + 256], v2 = src[c + 512];
    float mu = block_sum256(v0 + v1 + v2, red) * (1.f / DIMC);
    float d0 = v0 - mu, d1 = v1 - mu, d2 = v2 - mu;
    float var = block_sum256(d0 * d0 + d1 * d1 + d2 * d2, red) * (1.f / DIMC);
    float rs = rsqrtf(var + 1e-5f);
    dst[c]       = (__bf16)(d0 * rs * w[c]       + b[c]);
    dst[c + 256] = (__bf16)(d1 * rs * w[c + 256] + b[c + 256]);
    dst[c + 512] = (__bf16)(d2 * rs * w[c + 512] + b[c + 512]);
}

// LN2: rows of d_out (fp32) -> bf16
__global__ __launch_bounds__(256) void ln2_kernel(const float* __restrict__ xin,
                                                  const float* __restrict__ w,
                                                  const float* __restrict__ b,
                                                  __bf16* __restrict__ h2) {
    __shared__ float red[256];
    size_t row = blockIdx.x;
    const float* src = xin + row * DIMC;
    __bf16* dst = h2 + row * DIMC;
    int c = threadIdx.x;
    float v0 = src[c], v1 = src[c + 256], v2 = src[c + 512];
    float mu = block_sum256(v0 + v1 + v2, red) * (1.f / DIMC);
    float d0 = v0 - mu, d1 = v1 - mu, d2 = v2 - mu;
    float var = block_sum256(d0 * d0 + d1 * d1 + d2 * d2, red) * (1.f / DIMC);
    float rs = rsqrtf(var + 1e-5f);
    dst[c]       = (__bf16)(d0 * rs * w[c]       + b[c]);
    dst[c + 256] = (__bf16)(d1 * rs * w[c + 256] + b[c + 256]);
    dst[c + 512] = (__bf16)(d2 * rs * w[c + 512] + b[c + 512]);
}

// ---------------- generic WMMA GEMM: C = A(MxK) * B(KxN), epilogue fused ----------------
struct EpiBiasBf16 {            // QKV: out bf16, + bias
    const float* bias; __bf16* out; int ldo;
    __device__ void operator()(int gm, int gn, float v) const {
        out[(size_t)gm * ldo + gn] = (__bf16)(v + bias[gn]);
    }
};
struct EpiGelu {                // MLP1: + bias, exact GELU, bf16 out
    const float* bias; __bf16* out; int ldo;
    __device__ void operator()(int gm, int gn, float v) const {
        float t = v + bias[gn];
        float g = 0.5f * t * (1.0f + erff(t * 0.70710678118654752f));
        out[(size_t)gm * ldo + gn] = (__bf16)g;
    }
};
struct EpiProj {                // proj: + bias, window-unpartition, + residual x, fp32 out
    const float* bias; const float* xres; float* out;
    __device__ void operator()(int gm, int gn, float v) const {
        int win = gm / TOKP, tok = gm - win * TOKP;
        if (tok >= NTOK) return;
        int bb = win / 25, wr = win - bb * 25;
        int y  = (wr / 5) * WS + tok / WS;
        int xx = (wr % 5) * WS + tok % WS;
        if (y >= 64 || xx >= 64) return;
        size_t o = (((size_t)bb * 64 + y) * 64 + xx) * DIMC + gn;
        out[o] = xres[o] + v + bias[gn];
    }
};
struct EpiAddF32 {              // MLP2: + bias, accumulate into fp32 out (residual)
    const float* bias; float* out; int ldo;
    __device__ void operator()(int gm, int gn, float v) const {
        size_t o = (size_t)gm * ldo + gn;
        out[o] += v + bias[gn];
    }
};

// 128x128 macro tile, 256 threads = 8 waves, each wave 64x32 (4x2 WMMA tiles), K step 32.
// Software pipelined: global loads for tile i+1 issued into registers while tile i's
// fragments/WMMAs execute from LDS; loads are only waited on at the next LDS store.
template <class Epi>
__global__ __launch_bounds__(256) void gemm_bf16_kernel(const __bf16* __restrict__ A,
                                                        const __bf16* __restrict__ B,
                                                        int K, int ldb, Epi epi) {
    __shared__ __attribute__((aligned(32))) __bf16 As[128][32];   // [m][k]
    __shared__ __attribute__((aligned(32))) __bf16 Bs[128][32];   // [n][k] (transposed)
    int tid = threadIdx.x, lane = tid & 31, wave = tid >> 5;
    int mbase = blockIdx.y * 128, nbase = blockIdx.x * 128;
    int wm = (wave & 1) * 64, wn = (wave >> 1) * 32;

    v8f acc[4][2];
    #pragma unroll
    for (int i = 0; i < 4; i++)
        #pragma unroll
        for (int j = 0; j < 2; j++) acc[i][j] = v8f{};

    int arow = tid >> 1, ako = (tid & 1) * 16;          // A staging: 16 bf16 per thread
    int bk   = tid >> 3, bno = (tid & 7) * 16;          // B staging: 16 bf16 per thread
    int fr   = lane & 15;
    int fk8  = (lane >> 4) * 8;                          // unused; A frags also use fk16
    int fk16 = (lane >> 4) * 16;                         // B-frag K base

    const __bf16* aptr = A + (size_t)(mbase + arow) * K + ako;
    const __bf16* bptr = B + (size_t)bk * ldb + nbase + bno;

    // prologue: first tile into registers
    v8bf ra0 = *(const v8bf*)aptr,          ra1 = *(const v8bf*)(aptr + 8);
    v8bf rb0 = *(const v8bf*)bptr,          rb1 = *(const v8bf*)(bptr + 8);

    for (int k0 = 0; k0 < K; k0 += 32) {
        // stage current tile registers -> LDS
        *(v8bf*)&As[arow][ako]     = ra0;
        *(v8bf*)&As[arow][ako + 8] = ra1;
        #pragma unroll
        for (int i = 0; i < 8; i++) { Bs[bno + i][bk] = rb0[i]; Bs[bno + 8 + i][bk] = rb1[i]; }
        __syncthreads();

        // issue next tile's global loads (overlap with compute below)
        if (k0 + 32 < K) {
            const __bf16* an = aptr + k0 + 32;
            const __bf16* bn = bptr + (size_t)(k0 + 32) * ldb;
            ra0 = *(const v8bf*)an; ra1 = *(const v8bf*)(an + 8);
            rb0 = *(const v8bf*)bn; rb1 = *(const v8bf*)(bn + 8);
            if (k0 + 64 < K) __builtin_prefetch(aptr + k0 + 64, 0, 1);   // global_prefetch
        }

        // compute from LDS
        v16bf af[4], bfr[2];
        #pragma unroll
        for (int mi = 0; mi < 4; mi++)
            af[mi] = join8(*(const v8bf*)&As[wm + mi * 16 + fr][fk16],
                           *(const v8bf*)&As[wm + mi * 16 + fr][fk16 + 8]);
        #pragma unroll
        for (int ni = 0; ni < 2; ni++)
            bfr[ni] = *(const v16bf*)&Bs[wn + ni * 16 + fr][fk16];
        #pragma unroll
        for (int mi = 0; mi < 4; mi++)
            #pragma unroll
            for (int ni = 0; ni < 2; ni++)
                acc[mi][ni] = wmma_bf16(af[mi], bfr[ni], acc[mi][ni]);
        __syncthreads();
    }

    #pragma unroll
    for (int mi = 0; mi < 4; mi++)
        #pragma unroll
        for (int ni = 0; ni < 2; ni++)
            #pragma unroll
            for (int j = 0; j < 8; j++) {
                int gm = mbase + wm + mi * 16 + j + ((lane >> 4) << 3);
                int gn = nbase + wn + ni * 16 + (lane & 15);
                epi(gm, gn, acc[mi][ni][j]);
            }
}

// ---------------- flash attention: one block per (window, head) ----------------
// qkv: [200][224][2304] bf16  (q|k|v each [12][64] inside last dim)
// out: [200][224][768]  bf16  (dim = head*64 + hd)
__global__ __launch_bounds__(256) void attn_kernel(const __bf16* __restrict__ qkv,
                                                   const int* __restrict__ q_idx,
                                                   const int* __restrict__ k_idx,
                                                   const float* __restrict__ Sh,
                                                   const float* __restrict__ Sw,
                                                   __bf16* __restrict__ out) {
    __shared__ __attribute__((aligned(32))) __bf16 Ksh[TOKP][HD];     // K, TDM-staged row-major
    __shared__ __attribute__((aligned(32))) __bf16 Vsh[HD][TOKP];     // V transposed
    __shared__ __attribute__((aligned(32))) __bf16 Psh[8][16][32];    // per-wave P staging
    __shared__ int   qidx_s[NTOK], kidx_s[NTOK];
    __shared__ float Sh_s[NTOK], Sw_s[NTOK];

    int bh = blockIdx.x;                  // 0..2399
    int win = bh / NHEAD, head = bh - win * NHEAD;
    int tid = threadIdx.x, lane = tid & 31, wave = tid >> 5;
    const __bf16* base = qkv + (size_t)win * TOKP * (3 * DIMC);
    int fk8  = (lane >> 4) * 8;
    int fk16 = (lane >> 4) * 16;

    // ---- TDM: DMA the K tile (224 rows x 64 bf16, global stride 2304 elems) into Ksh ----
    if (wave == 0) {
        uint64_t gaddr = (uint64_t)(uintptr_t)(base + DIMC + head * HD);   // K slice start (bytes)
        uint32_t laddr = (uint32_t)(uintptr_t)(&Ksh[0][0]);                // LDS byte offset
        u32x4 g0 = {};
        u32x8 g1 = {};
        u32x4 g2 = {};
        u32x4 g3 = {};
        g0[0] = 1u;                                             // count=1 (valid descriptor)
        g0[1] = laddr;                                          // lds_addr [63:32]
        g0[2] = (uint32_t)gaddr;                                // global_addr [95:64]
        g0[3] = (uint32_t)((gaddr >> 32) & 0x01FFFFFFu)         // global_addr [120:96]
              | (2u << 30);                                     // type=2 ("image") [127:126]
        g1[0] = 1u << 16;                                       // data_size=1 (2 bytes)
        g1[1] = (uint32_t)HD << 16;                             // tensor_dim0 (low16 @ bits63:48)
        g1[2] = (uint32_t)TOKP << 16;                           // tensor_dim1 (low16 @ bits111:80)
        g1[3] = (uint32_t)HD << 16;                             // tile_dim0 @ bits127:112
        g1[4] = (uint32_t)TOKP;                                 // tile_dim1 @ bits143:128
        g1[5] = (uint32_t)(3 * DIMC);                           // tensor_dim0_stride (elements)
        asm volatile("tensor_load_to_lds %0, %1, %2, %3"
                     :: "s"(g0), "s"(g1), "s"(g2), "s"(g3) : "memory");
    }

    // stage V transposed into LDS (manual: TDM cannot transpose)
    for (int idx = tid; idx < TOKP * 4; idx += 256) {
        int tok = idx >> 2, ho = (idx & 3) * 16;
        const __bf16* src = base + (size_t)tok * (3 * DIMC) + 2 * DIMC + head * HD + ho;
        v8bf x0 = *(const v8bf*)src, x1 = *(const v8bf*)(src + 8);
        #pragma unroll
        for (int i = 0; i < 8; i++) { Vsh[ho + i][tok] = x0[i]; Vsh[ho + 8 + i][tok] = x1[i]; }
    }
    for (int i = tid; i < NTOK; i += 256) {
        qidx_s[i] = q_idx[(size_t)bh * NTOK + i];
        kidx_s[i] = k_idx[(size_t)bh * NTOK + i];
        Sh_s[i] = Sh[i]; Sw_s[i] = Sw[i];
    }
    if (wave == 0) wait_tensorcnt0();     // K DMA complete (wave 0 owns TENSORcnt)
    __syncthreads();                      // publish Ksh/Vsh/tables to all waves

    const float scale = 0.125f;   // HD^-0.5
    for (int qt = wave; qt < TOKP / 16; qt += 8) {
        // q A-fragments (16 rows x 64 hd -> two 16x32 frags), straight from global
        int qtok = qt * 16 + (lane & 15);
        const __bf16* qrow = base + (size_t)qtok * (3 * DIMC) + head * HD;
        v16bf aq[2];
        #pragma unroll
        for (int f = 0; f < 2; f++)
            aq[f] = join8(*(const v8bf*)(qrow + f * 32 + fk16),
                          *(const v8bf*)(qrow + f * 32 + fk16 + 8));

        // hoist q-side rel-pos decomposition out of the k loop (branchless, clamped gather)
        int   qr14[8], qc14[8];
        float qok[8];
        #pragma unroll
        for (int j = 0; j < 8; j++) {
            int gq = qt * 16 + j + ((lane >> 4) << 3);
            int qi = qidx_s[gq < NTOK ? gq : NTOK - 1];
            qr14[j] = (qi / WS) * WS;
            qc14[j] = (qi % WS) * WS;
            qok[j]  = (gq < NTOK) ? 1.0f : 0.0f;
        }

        float m_i[8], l_i[8];
        v8f O[4];
        #pragma unroll
        for (int j = 0; j < 8; j++) { m_i[j] = -1e30f; l_i[j] = 0.f; }
        #pragma unroll
        for (int h = 0; h < 4; h++) O[h] = v8f{};

        for (int kt = 0; kt < TOKP / 32; kt++) {
            float ps[2][8];
            #pragma unroll
            for (int c = 0; c < 2; c++) {
                int n = kt * 32 + c * 16 + (lane & 15);           // key token (B-frag column)
                v8f s = v8f{};
                s = wmma_bf16(aq[0], *(const v16bf*)&Ksh[n][fk16], s);        // hd 0..31
                s = wmma_bf16(aq[1], *(const v16bf*)&Ksh[n][32 + fk16], s);   // hd 32..63
                // k-side rel-pos decomposition hoisted out of the j loop
                int ki  = kidx_s[n < NTOK ? n : NTOK - 1];
                int kr  = ki / WS, kc = ki % WS;
                bool kok = (n < NTOK);
                #pragma unroll
                for (int j = 0; j < 8; j++) {
                    float bias = Sh_s[qr14[j] + kr] + Sw_s[qc14[j] + kc];
                    float v = s[j] * scale + bias;
                    ps[c][j] = (kok && qok[j] > 0.f) ? v : -1e30f;   // v_cndmask, no exec dance
                }
            }
            // online softmax (row stats across 16 lanes of each half)
            #pragma unroll
            for (int j = 0; j < 8; j++) {
                float rmax = fmaxf(ps[0][j], ps[1][j]);
                #pragma unroll
                for (int d = 8; d > 0; d >>= 1) rmax = fmaxf(rmax, __shfl_xor(rmax, d, 32));
                float nm = fmaxf(m_i[j], rmax);
                float corr = __expf(m_i[j] - nm);
                float p0 = __expf(ps[0][j] - nm), p1 = __expf(ps[1][j] - nm);
                float rsum = p0 + p1;
                #pragma unroll
                for (int d = 8; d > 0; d >>= 1) rsum += __shfl_xor(rsum, d, 32);
                m_i[j] = nm;
                l_i[j] = l_i[j] * corr + rsum;
                #pragma unroll
                for (int h = 0; h < 4; h++) O[h][j] *= corr;
                int row = j + ((lane >> 4) << 3);
                Psh[wave][row][lane & 15]        = (__bf16)p0;    // C layout -> LDS
                Psh[wave][row][16 + (lane & 15)] = (__bf16)p1;
            }
            asm volatile("s_wait_dscnt 0" ::: "memory");          // wave-local P visibility
            v16bf ap = join8(*(const v8bf*)&Psh[wave][lane & 15][fk16],
                             *(const v8bf*)&Psh[wave][lane & 15][fk16 + 8]);
            #pragma unroll
            for (int h = 0; h < 4; h++) {
                int n = h * 16 + (lane & 15);                     // hd column
                O[h] = wmma_bf16(ap, *(const v16bf*)&Vsh[n][kt * 32 + fk16], O[h]);
            }
        }
        // normalize + store
        #pragma unroll
        for (int j = 0; j < 8; j++) {
            float inv = 1.0f / l_i[j];
            #pragma unroll
            for (int h = 0; h < 4; h++) O[h][j] *= inv;
        }
        #pragma unroll
        for (int h = 0; h < 4; h++)
            #pragma unroll
            for (int j = 0; j < 8; j++) {
                int tok = qt * 16 + j + ((lane >> 4) << 3);
                int hd  = h * 16 + (lane & 15);
                out[((size_t)win * TOKP + tok) * DIMC + head * HD + hd] = (__bf16)O[h][j];
            }
    }
}

// ---------------- launch ----------------
extern "C" void kernel_launch(void* const* d_in, const int* in_sizes, int n_in,
                              void* d_out, int out_size, void* d_ws, size_t ws_size,
                              hipStream_t stream) {
    const float* x      = (const float*)d_in[0];
    const int*   q_idx  = (const int*)d_in[1];
    const int*   k_idx  = (const int*)d_in[2];
    const float* ln1_w  = (const float*)d_in[3];
    const float* ln1_b  = (const float*)d_in[4];
    const float* ln2_w  = (const float*)d_in[5];
    const float* ln2_b  = (const float*)d_in[6];
    const float* qkv_w  = (const float*)d_in[7];
    const float* qkv_b  = (const float*)d_in[8];
    const float* proj_w = (const float*)d_in[9];
    const float* proj_b = (const float*)d_in[10];
    const float* rel_h  = (const float*)d_in[11];
    const float* rel_w  = (const float*)d_in[12];
    const float* mlp_w1 = (const float*)d_in[13];
    const float* mlp_b1 = (const float*)d_in[14];
    const float* mlp_w2 = (const float*)d_in[15];
    const float* mlp_b2 = (const float*)d_in[16];
    float* out = (float*)d_out;

    // workspace cursor
    char* cur = (char*)d_ws;
    auto alloc = [&](size_t bytes) {
        char* p = cur;
        cur += (bytes + 255) & ~(size_t)255;
        return (void*)p;
    };
    float*  tSh   = (float*)alloc(NTOK * 4);
    float*  tSw   = (float*)alloc(NTOK * 4);
    __bf16* wqkv  = (__bf16*)alloc((size_t)DIMC * 3 * DIMC * 2);
    __bf16* wproj = (__bf16*)alloc((size_t)DIMC * DIMC * 2);
    __bf16* wm1   = (__bf16*)alloc((size_t)DIMC * 4 * DIMC * 2);
    __bf16* wm2   = (__bf16*)alloc((size_t)4 * DIMC * DIMC * 2);
    __bf16* hwin  = (__bf16*)alloc((size_t)MWIN * DIMC * 2);       // LN1 windows; reused as attn out
    __bf16* qkvb  = (__bf16*)alloc((size_t)MWIN * 3 * DIMC * 2);   // QKV; reused as MLP hidden
    __bf16* h2    = (__bf16*)alloc((size_t)MTOK * DIMC * 2);       // LN2 out
    __bf16* attno = hwin;                                           // alias (hwin consumed by QKV gemm)
    __bf16* m1    = qkvb;                                           // alias (qkv consumed by attention)

    // prep: rel-pos tables + weight conversion
    rel_table_kernel<<<2, 256, 0, stream>>>(rel_h, rel_w, tSh, tSw);
    int nq = DIMC * 3 * DIMC, np = DIMC * DIMC, nm = DIMC * 4 * DIMC;
    cvt_bf16_kernel<<<(nq + 255) / 256, 256, 0, stream>>>(qkv_w,  wqkv,  nq);
    cvt_bf16_kernel<<<(np + 255) / 256, 256, 0, stream>>>(proj_w, wproj, np);
    cvt_bf16_kernel<<<(nm + 255) / 256, 256, 0, stream>>>(mlp_w1, wm1,  nm);
    cvt_bf16_kernel<<<(nm + 255) / 256, 256, 0, stream>>>(mlp_w2, wm2,  nm);

    // LN1 + window partition
    ln1_win_kernel<<<MWIN, 256, 0, stream>>>(x, ln1_w, ln1_b, hwin);

    // QKV GEMM: [44800,768] x [768,2304]
    gemm_bf16_kernel<EpiBiasBf16><<<dim3(3 * DIMC / 128, MWIN / 128), 256, 0, stream>>>(
        hwin, wqkv, DIMC, 3 * DIMC, EpiBiasBf16{qkv_b, qkvb, 3 * DIMC});

    // attention per (window, head)
    attn_kernel<<<NWIN * NHEAD, 256, 0, stream>>>(qkvb, q_idx, k_idx, tSh, tSw, attno);

    // proj GEMM + unpartition + residual -> d_out (fp32)
    gemm_bf16_kernel<EpiProj><<<dim3(DIMC / 128, MWIN / 128), 256, 0, stream>>>(
        attno, wproj, DIMC, DIMC, EpiProj{proj_b, x, out});

    // LN2
    ln2_kernel<<<MTOK, 256, 0, stream>>>(out, ln2_w, ln2_b, h2);

    // MLP1 + GELU: [32768,768] x [768,3072]
    gemm_bf16_kernel<EpiGelu><<<dim3(4 * DIMC / 128, MTOK / 128), 256, 0, stream>>>(
        h2, wm1, DIMC, 4 * DIMC, EpiGelu{mlp_b1, m1, 4 * DIMC});

    // MLP2 + residual add into d_out: [32768,3072] x [3072,768]
    gemm_bf16_kernel<EpiAddF32><<<dim3(DIMC / 128, MTOK / 128), 256, 0, stream>>>(
        m1, wm2, 4 * DIMC, DIMC, EpiAddF32{mlp_b2, out, DIMC});

    (void)in_sizes; (void)n_in; (void)out_size; (void)ws_size;
}
